// GraphAttentionLayer_37314675868405
// MI455X (gfx1250) — matrix operand order repaired and checked by reading, not compiled, not run on hardware
//
#include <hip/hip_runtime.h>
#include <hip/hip_bf16.h>

typedef __attribute__((ext_vector_type(16))) _Float16 v16h;
typedef __attribute__((ext_vector_type(8)))  _Float16 v8h;
typedef __attribute__((ext_vector_type(8)))  float    v8f;

#define N_NODES 8192
#define IN_DIM  512
#define OUT_DIM 64
#define SLOPE   0.2f
#define JCHUNK  128
#define MASK_NEG (-1.0e9f)

// ---------------------------------------------------------------------------
// Kernel 1: Wh = x @ W^T (f32 GEMM, 64x64 output tile per block), writes
//   - WhT  : f16 transposed [OUT_DIM][N_NODES]  (WMMA B-operand-ready)
//   - s[i] = Wh[i,:] . a_src   (exact f32, deterministic tree reduce)
//   - d[i] = Wh[i,:] . a_dst
// ---------------------------------------------------------------------------
__global__ __launch_bounds__(256)
void gat_proj_kernel(const float* __restrict__ x, const float* __restrict__ W,
                     const float* __restrict__ a_src, const float* __restrict__ a_dst,
                     _Float16* __restrict__ WhT, float* __restrict__ s_out,
                     float* __restrict__ d_out) {
    __shared__ float xs[64][33];
    __shared__ float wsh[64][33];
    __shared__ float red[64][16];

    const int tid = threadIdx.x;
    const int tx  = tid & 15;   // col group (4 cols each)
    const int ty  = tid >> 4;   // row group (4 rows each)
    const int i0  = blockIdx.x * 64;

    float acc[4][4];
#pragma unroll
    for (int j = 0; j < 4; ++j)
#pragma unroll
        for (int jj = 0; jj < 4; ++jj) acc[j][jj] = 0.f;

    const int lrow = tid >> 2;        // 0..63
    const int lk   = (tid & 3) * 8;   // 0,8,16,24

    for (int kc = 0; kc < IN_DIM; kc += 32) {
        const float* xp = &x[(size_t)(i0 + lrow) * IN_DIM + kc + lk];
        float4 xa = *(const float4*)xp;
        float4 xb = *(const float4*)(xp + 4);
        const float* wp = &W[(size_t)lrow * IN_DIM + kc + lk];
        float4 wa = *(const float4*)wp;
        float4 wb = *(const float4*)(wp + 4);
        xs[lrow][lk + 0] = xa.x; xs[lrow][lk + 1] = xa.y;
        xs[lrow][lk + 2] = xa.z; xs[lrow][lk + 3] = xa.w;
        xs[lrow][lk + 4] = xb.x; xs[lrow][lk + 5] = xb.y;
        xs[lrow][lk + 6] = xb.z; xs[lrow][lk + 7] = xb.w;
        wsh[lrow][lk + 0] = wa.x; wsh[lrow][lk + 1] = wa.y;
        wsh[lrow][lk + 2] = wa.z; wsh[lrow][lk + 3] = wa.w;
        wsh[lrow][lk + 4] = wb.x; wsh[lrow][lk + 5] = wb.y;
        wsh[lrow][lk + 6] = wb.z; wsh[lrow][lk + 7] = wb.w;
        __syncthreads();
#pragma unroll 8
        for (int k = 0; k < 32; ++k) {
            float av[4], bv[4];
#pragma unroll
            for (int j = 0; j < 4; ++j) av[j] = xs[ty * 4 + j][k];
#pragma unroll
            for (int jj = 0; jj < 4; ++jj) bv[jj] = wsh[tx * 4 + jj][k];
#pragma unroll
            for (int j = 0; j < 4; ++j)
#pragma unroll
                for (int jj = 0; jj < 4; ++jj)
                    acc[j][jj] = fmaf(av[j], bv[jj], acc[j][jj]);
        }
        __syncthreads();
    }

    // Transposed f16 store: WhT[col][row]
#pragma unroll
    for (int j = 0; j < 4; ++j)
#pragma unroll
        for (int jj = 0; jj < 4; ++jj)
            WhT[(size_t)(tx * 4 + jj) * N_NODES + (i0 + ty * 4 + j)] =
                (_Float16)acc[j][jj];

    // s reduction (deterministic)
#pragma unroll
    for (int j = 0; j < 4; ++j) {
        float ps = 0.f;
#pragma unroll
        for (int jj = 0; jj < 4; ++jj) ps = fmaf(acc[j][jj], a_src[tx * 4 + jj], ps);
        red[ty * 4 + j][tx] = ps;
    }
    __syncthreads();
    if (tid < 64) {
        float t = 0.f;
#pragma unroll
        for (int g = 0; g < 16; ++g) t += red[tid][g];
        s_out[i0 + tid] = t;
    }
    __syncthreads();
    // d reduction
#pragma unroll
    for (int j = 0; j < 4; ++j) {
        float pd = 0.f;
#pragma unroll
        for (int jj = 0; jj < 4; ++jj) pd = fmaf(acc[j][jj], a_dst[tx * 4 + jj], pd);
        red[ty * 4 + j][tx] = pd;
    }
    __syncthreads();
    if (tid < 64) {
        float t = 0.f;
#pragma unroll
        for (int g = 0; g < 16; ++g) t += red[tid][g];
        d_out[i0 + tid] = t;
    }
}

// ---------------------------------------------------------------------------
// Kernel 2: dmax = max_j d[j] (single block)
// ---------------------------------------------------------------------------
__global__ __launch_bounds__(1024)
void gat_dmax_kernel(const float* __restrict__ d, float* __restrict__ dmax) {
    __shared__ float red[1024];
    float m = -3.0e38f;
    for (int i = threadIdx.x; i < N_NODES; i += 1024) m = fmaxf(m, d[i]);
    red[threadIdx.x] = m;
    __syncthreads();
    for (int st = 512; st > 0; st >>= 1) {
        if (threadIdx.x < st)
            red[threadIdx.x] = fmaxf(red[threadIdx.x], red[threadIdx.x + st]);
        __syncthreads();
    }
    if (threadIdx.x == 0) *dmax = red[0];
}

// ---------------------------------------------------------------------------
// Kernel 3: fused masked-softmax attention + aggregation.
// Block = 16 output rows, 128 threads (4 waves). Streams adj row-block once.
// Chunk = 128 j's per iteration: P[16x128] f16 in LDS (ping-pong buffer, one
// barrier per iteration), then each wave issues 4 back-to-back
// v_wmma_f32_16x16x32_f16 into its 16x16 output tile (4 waves = 64 cols).
// Masking is branch-free: the exp *input* is selected (-1e9 when masked), so
// v_exp_f32 underflows to exactly 0 — same semantics as the reference NEG_INF.
// m_i = leakyrelu(s_i + max_j d_j) upper-bounds the masked row max (LeakyReLU
// is monotone), so exp <= 1: no overflow, no online rescaling, one pass.
// ---------------------------------------------------------------------------
__global__ __launch_bounds__(128)
void gat_attn_kernel(const int* __restrict__ adj, const _Float16* __restrict__ WhT,
                     const float* __restrict__ s, const float* __restrict__ d,
                     const float* __restrict__ dmaxp, float* __restrict__ out) {
    __shared__ float s_sh[16];
    __shared__ float m_sh[16];
    __shared__ __align__(32) _Float16 Psh[2][16 * JCHUNK];   // 2 x 4KB ping-pong
    __shared__ float red16[16][8];
    __shared__ float linv[16];

    const int tid  = threadIdx.x;
    const int lane = tid & 31;
    const int wave = tid >> 5;
    const int i0   = blockIdx.x * 16;

    if (tid < 16) {
        float sv = s[i0 + tid];
        float e  = sv + *dmaxp;
        s_sh[tid] = sv;
        m_sh[tid] = fmaxf(e, SLOPE * e);   // leakyrelu upper bound (slope<1)
    }
    __syncthreads();

    // P-computation mapping: thread owns row r, 16 consecutive cols jb..jb+15
    const int r  = tid >> 3;          // 0..15
    const int jb = (tid & 7) * 16;    // 0..112
    const float  sr   = s_sh[r];
    const float  mr   = m_sh[r];
    const size_t arow = (size_t)(i0 + r) * N_NODES;
    float lpriv = 0.f;

    // WMMA fragment mapping
    const int colg = 16 * wave + (lane & 15);   // output column of this lane
    const _Float16* bbase = WhT + (size_t)colg * N_NODES + ((lane < 16) ? 0 : 16);
    const int am    = (lane < 16) ? lane : (lane - 16);
    const int aoff0 = am * JCHUNK + ((lane < 16) ? 0 : 16);   // K 0-7 / 16-23
    const int aoff1 = am * JCHUNK + ((lane < 16) ? 8 : 24);   // K 8-15 / 24-31

    v8f acc = {};

    // Software pipeline: preload chunk 0
    int4   a_reg[4];
    float4 d_reg[4];
#pragma unroll
    for (int q = 0; q < 4; ++q) {
        a_reg[q] = *(const int4*)&adj[arow + jb + q * 4];
        d_reg[q] = *(const float4*)&d[jb + q * 4];
    }

    for (int j0 = 0, it = 0; j0 < N_NODES; j0 += JCHUNK, ++it) {
        const int buf = it & 1;

        // Compute 16 attention weights, branch-free (select exp input)
        v8h pha, phb;
#pragma unroll
        for (int q = 0; q < 4; ++q) {
            float dv[4] = {d_reg[q].x, d_reg[q].y, d_reg[q].z, d_reg[q].w};
            int   av[4] = {a_reg[q].x, a_reg[q].y, a_reg[q].z, a_reg[q].w};
#pragma unroll
            for (int k = 0; k < 4; ++k) {
                float e  = sr + dv[k];
                float lr = fmaxf(e, SLOPE * e);            // leakyrelu, 1 VALU
                float t  = (av[k] > 0) ? (lr - mr) : MASK_NEG;
                _Float16 ph = (_Float16)__expf(t);         // exp(-1e9) == 0
                lpriv += (float)ph;        // denominator matches f16 numerator
                int idx = q * 4 + k;
                if (idx < 8) pha[idx] = ph; else phb[idx - 8] = ph;
            }
        }
        *(v8h*)&Psh[buf][r * JCHUNK + jb]     = pha;
        *(v8h*)&Psh[buf][r * JCHUNK + jb + 8] = phb;

        // Issue next chunk's loads (overlap barrier + WMMAs with HBM latency)
        const int jn = j0 + JCHUNK;
        if (jn < N_NODES) {
#pragma unroll
            for (int q = 0; q < 4; ++q) {
                a_reg[q] = *(const int4*)&adj[arow + jn + jb + q * 4];
                d_reg[q] = *(const float4*)&d[jn + jb + q * 4];
            }
            if (jn + JCHUNK < N_NODES)
                __builtin_prefetch(&adj[arow + jn + JCHUNK + jb], 0, 0);
        }

        __syncthreads();   // single barrier: ping-pong removes the WAR hazard

        // 4 back-to-back WMMAs over the 128-wide chunk
#pragma unroll
        for (int kb = 0; kb < 4; ++kb) {
            v8h lo = *(const v8h*)&Psh[buf][aoff0 + kb * 32];
            v8h hi = *(const v8h*)&Psh[buf][aoff1 + kb * 32];
            v16h A;
#pragma unroll
            for (int t = 0; t < 8; ++t) { A[t] = lo[t]; A[t + 8] = hi[t]; }
            v16h B = *(const v16h*)&bbase[j0 + kb * 32];
            acc = __builtin_amdgcn_wmma_f32_16x16x32_f16(
                /*neg_a=*/false, A, /*neg_b=*/false, B,
                /*c_mod=*/(short)0, acc, /*reuse_a=*/false, /*reuse_b=*/false);
        }
    }

    // Row-sum reduction (deterministic) and normalization
    red16[r][tid & 7] = lpriv;
    __syncthreads();
    if (tid < 16) {
        float t = 0.f;
#pragma unroll
        for (int g = 0; g < 8; ++g) t += red16[tid][g];
        linv[tid] = (t > 0.f) ? 1.0f / t : 0.f;
    }
    __syncthreads();

    const int rbase = (lane < 16) ? 0 : 8;   // C/D layout: VGPR v -> M=v or v+8
#pragma unroll
    for (int v = 0; v < 8; ++v) {
        int row = i0 + v + rbase;
        out[(size_t)row * OUT_DIM + colg] = acc[v] * linv[v + rbase];
    }
}

// ---------------------------------------------------------------------------
extern "C" void kernel_launch(void* const* d_in, const int* in_sizes, int n_in,
                              void* d_out, int out_size, void* d_ws, size_t ws_size,
                              hipStream_t stream) {
    const float* x     = (const float*)d_in[0];
    const int*   adj   = (const int*)d_in[1];
    const float* W     = (const float*)d_in[2];
    const float* a_src = (const float*)d_in[3];
    const float* a_dst = (const float*)d_in[4];
    float* out = (float*)d_out;

    char* ws = (char*)d_ws;
    _Float16* WhT = (_Float16*)ws;                                   // 1 MB
    float* s    = (float*)(ws + (size_t)OUT_DIM * N_NODES * sizeof(_Float16));
    float* d    = s + N_NODES;
    float* dmax = d + N_NODES;

    gat_proj_kernel<<<N_NODES / 64, 256, 0, stream>>>(x, W, a_src, a_dst, WhT, s, d);
    gat_dmax_kernel<<<1, 1024, 0, stream>>>(d, dmax);
    gat_attn_kernel<<<N_NODES / 16, 128, 0, stream>>>(adj, WhT, s, d, dmax, out);
}
